// LC_Mamba_Block_88596585381962
// MI455X (gfx1250) — matrix operand order repaired and checked by reading, not compiled, not run on hardware
//
#include <hip/hip_runtime.h>
#include <math.h>

typedef __attribute__((ext_vector_type(16))) _Float16 v16h;
typedef __attribute__((ext_vector_type(8)))  float    v8f;

#define Hh 192
#define Ww 192
#define HW (Hh*Ww)
#define SHIFTs 4

__device__ __forceinline__ float warp_sum(float v) {
#pragma unroll
  for (int m = 16; m >= 1; m >>= 1) v += __shfl_xor(v, m, 32);
  return v;
}
__device__ __forceinline__ float siluf(float x)     { return x / (1.f + __expf(-x)); }
__device__ __forceinline__ float sigm_(float x)     { return 1.f / (1.f + __expf(-x)); }
__device__ __forceinline__ float softplus_(float x) { return (x > 20.f) ? x : log1pf(__expf(x)); }

// A-fragment (16x32 f16): lane holds row M=lane%16; halves = 16 consecutive K.
__device__ __forceinline__ v16h fragA_lds(const _Float16* tile, int row, int k0, int hi, int rs) {
  union { v16h v; _Float16 h[16]; } u;
  const _Float16* rp = tile + row * rs;
  int klo = k0 + (hi ? 16 : 0);
#pragma unroll
  for (int i = 0; i < 16; ++i) { u.h[i] = rp[klo + i]; }
  return u.v;
}
// B-fragment (32x16 f16): lane holds col N=lane%16; halves = 16 consecutive K.
// tile stored transposed [n][k] in LDS.
__device__ __forceinline__ v16h fragB_lds(const _Float16* tile, int n, int k0, int hi, int rs) {
  union { v16h v; _Float16 h[16]; } u;
  const _Float16* rp = tile + n * rs + k0 + (hi ? 16 : 0);
#pragma unroll
  for (int i = 0; i < 16; ++i) u.h[i] = rp[i];
  return u.v;
}
// B-fragment from global f32 row-major [k][n] (ld = row stride); zero-pads n>=nmax.
__device__ __forceinline__ v16h fragB_glb(const float* B, int n, int k0, int hi, int ld, int nmax) {
  union { v16h v; _Float16 h[16]; } u;
  int kb = k0 + (hi ? 16 : 0);
  if (n < nmax) {
#pragma unroll
    for (int i = 0; i < 16; ++i) u.h[i] = (_Float16)B[(kb + i) * ld + n];
  } else {
#pragma unroll
    for (int i = 0; i < 16; ++i) u.h[i] = (_Float16)0.f;
  }
  return u.v;
}

// ---------------- K0: NCHW -> NHWC ----------------
__global__ void k0_nchw_to_nhwc(const float* __restrict__ x, float* __restrict__ sc) {
  __shared__ float t[64][65];
  int b = blockIdx.y;
  int base = blockIdx.x * 64;
  int tid = threadIdx.x;
  int pix = tid & 63, cq = tid >> 6;
  for (int c = cq; c < 64; c += 4)
    t[pix][c] = x[(size_t)(b * 64 + c) * HW + base + pix];
  __syncthreads();
  int c2 = tid & 63, pq = tid >> 6;
  for (int p = pq; p < 64; p += 4)
    sc[((size_t)b * HW + base + p) * 64 + c2] = t[p][c2];
}

// ---------------- K1: shift + LN1 + in-projection (WMMA) ----------------
__global__ void k1_ln_inproj(const float* __restrict__ sc,
                             const float* __restrict__ ln_g, const float* __restrict__ ln_b,
                             const float* __restrict__ in_w, const float* __restrict__ in_b,
                             float* __restrict__ xin, float* __restrict__ zbuf) {
  __shared__ _Float16 At[64 * 72];     // xn tile, f16, [p][k]
  __shared__ _Float16 Bt[256 * 72];    // in_w transposed, [n][k]
  int win = blockIdx.x;
  int b = win / 576, rem = win % 576;
  int wy = rem / 24, wx = rem % 24;
  int tid = threadIdx.x, lane = tid & 31, wid = tid >> 5;

  for (int idx = tid; idx < 64 * 256; idx += 256) {
    int k = idx >> 8, n = idx & 255;
    Bt[n * 72 + k] = (_Float16)in_w[idx];
  }
  float g0 = ln_g[lane], g1 = ln_g[lane + 32];
  float bb0 = ln_b[lane], bb1 = ln_b[lane + 32];
  for (int p = wid; p < 64; p += 8) {
    int hh = wy * 8 + (p >> 3), ww = wx * 8 + (p & 7);
    int sh = (hh + SHIFTs) % Hh, sw = (ww + SHIFTs) % Ww;
    const float* src = sc + ((size_t)b * HW + sh * Ww + sw) * 64;
    float v0 = src[lane], v1 = src[lane + 32];
    float s  = warp_sum(v0 + v1);
    float sq = warp_sum(v0 * v0 + v1 * v1);
    float mu = s * (1.f / 64.f);
    float var = sq * (1.f / 64.f) - mu * mu;
    float rs = rsqrtf(var + 1e-5f);
    At[p * 72 + lane]      = (_Float16)((v0 - mu) * rs * g0 + bb0);
    At[p * 72 + lane + 32] = (_Float16)((v1 - mu) * rs * g1 + bb1);
  }
  __syncthreads();

  int mtile = wid & 3, sel = wid >> 2;
  int hi = (lane & 16) ? 1 : 0;
  int mrow = mtile * 16 + (lane & 15);
#pragma unroll
  for (int j = 0; j < 8; ++j) {
    int ntile = sel + 2 * j;
    int ncol = ntile * 16 + (lane & 15);
    v8f acc = {0.f,0.f,0.f,0.f,0.f,0.f,0.f,0.f};
#pragma unroll
    for (int kc = 0; kc < 2; ++kc) {
      v16h a  = fragA_lds(At, mrow, kc * 32, hi, 72);
      v16h bf = fragB_lds(Bt, ncol, kc * 32, hi, 72);
      acc = __builtin_amdgcn_wmma_f32_16x16x32_f16(false, a, false, bf, (short)0, acc, false, false);
    }
    float bias = in_b[ncol];
#pragma unroll
    for (int r = 0; r < 8; ++r) {
      int m = mtile * 16 + (hi ? 8 : 0) + r;
      float val = acc[r] + bias;
      if (ncol < 128) {
        int hh = wy * 8 + (m >> 3), ww = wx * 8 + (m & 7);
        float msk = (hh < Hh - SHIFTs && ww < Ww - SHIFTs) ? 1.f : 0.f;
        xin[((size_t)win * 64 + m) * 128 + ncol] = val * msk;
      } else {
        zbuf[((size_t)win * 64 + m) * 128 + (ncol - 128)] = val;
      }
    }
  }
}

// ---------------- K2: conv -> x-proj (WMMA) -> scan -> out-proj (WMMA) + skip ----------------
__global__ void k2_conv_scan_outproj(
    const float* __restrict__ xin, const float* __restrict__ zbuf,
    const float* __restrict__ conv_w, const float* __restrict__ conv_b,
    const float* __restrict__ xproj_w,
    const float* __restrict__ dt_w, const float* __restrict__ dt_b,
    const float* __restrict__ A_log, const float* __restrict__ Dvec,
    const float* __restrict__ out_w, const float* __restrict__ out_b,
    const float* __restrict__ sc, const float* __restrict__ skip1,
    float* __restrict__ x1) {
  __shared__ float   xcf[64 * 128];     // xc, f32
  __shared__ _Float16 yf[64 * 136];     // xc f16, later y f16
  __shared__ float   dbc[64 * 48];      // dbc (36 used, padded)
  int win = blockIdx.x;
  int b = win / 576, rem = win % 576;
  int wy = rem / 24, wx = rem % 24;
  int tid = threadIdx.x, lane = tid & 31, wid = tid >> 5;
  const float* xw = xin + (size_t)win * 64 * 128;

  // causal depthwise conv (K=4) + SiLU
  for (int idx = tid; idx < 8192; idx += 256) {
    int p = idx >> 7, d = idx & 127;
    float acc = conv_b[d];
#pragma unroll
    for (int k = 0; k < 4; ++k) {
      int q = p - 3 + k;
      if (q >= 0) acc += xw[q * 128 + d] * conv_w[k * 128 + d];
    }
    float s = siluf(acc);
    xcf[idx] = s;
    yf[p * 136 + d] = (_Float16)s;
  }
  __syncthreads();

  // GEMM1: dbc = xc @ xproj_w  [64x128]@[128x36]
  int hi = (lane & 16) ? 1 : 0;
  for (int t = wid; t < 12; t += 8) {
    int mtile = t & 3, ntile = t >> 2;
    int ncol = ntile * 16 + (lane & 15);
    int mrow = mtile * 16 + (lane & 15);
    v8f acc = {0.f,0.f,0.f,0.f,0.f,0.f,0.f,0.f};
#pragma unroll
    for (int kc = 0; kc < 4; ++kc) {
      v16h a  = fragA_lds(yf, mrow, kc * 32, hi, 136);
      v16h bf = fragB_glb(xproj_w, ncol, kc * 32, hi, 36, 36);
      acc = __builtin_amdgcn_wmma_f32_16x16x32_f16(false, a, false, bf, (short)0, acc, false, false);
    }
#pragma unroll
    for (int r = 0; r < 8; ++r) {
      int m = mtile * 16 + (hi ? 8 : 0) + r;
      dbc[m * 48 + ncol] = acc[r];
    }
  }
  __syncthreads();

  // selective scan: 2 lanes per channel d, 8 states each; y combined by shfl_xor(1)
  {
    int d = wid * 16 + (lane >> 1);
    int shalf = (lane & 1) * 8;
    float At[8], hst[8];
#pragma unroll
    for (int j = 0; j < 8; ++j) { At[j] = -__expf(A_log[d * 16 + shalf + j]); hst[j] = 0.f; }
    float w0 = dt_w[d], w1 = dt_w[128 + d], w2 = dt_w[256 + d], w3 = dt_w[384 + d];
    float dtb = dt_b[d], Dd = Dvec[d];
    const float* zg = zbuf + (size_t)win * 64 * 128 + d;
    for (int p = 0; p < 64; ++p) {
      const float* row = dbc + p * 48;
      float dtv = softplus_(row[0]*w0 + row[1]*w1 + row[2]*w2 + row[3]*w3 + dtb);
      float xt = xcf[p * 128 + d];
      float coef = dtv * xt;
      float yp = 0.f;
#pragma unroll
      for (int j = 0; j < 8; ++j) {
        hst[j] = __expf(dtv * At[j]) * hst[j] + coef * row[4 + shalf + j];
        yp += hst[j] * row[20 + shalf + j];
      }
      float y = yp + __shfl_xor(yp, 1, 32);
      float zv = zg[p * 128];
      float yv = (y + Dd * xt) * siluf(zv);
      if ((lane & 1) == 0) yf[p * 136 + d] = (_Float16)yv;
    }
  }
  __syncthreads();

  // GEMM2: out = y @ out_w [64x128]@[128x64] + out_b; fused window-reverse + skip1
  for (int t = wid * 2; t < wid * 2 + 2; ++t) {
    int mtile = t >> 2, ntile = t & 3;
    int ncol = ntile * 16 + (lane & 15);
    int mrow = mtile * 16 + (lane & 15);
    v8f acc = {0.f,0.f,0.f,0.f,0.f,0.f,0.f,0.f};
#pragma unroll
    for (int kc = 0; kc < 4; ++kc) {
      v16h a  = fragA_lds(yf, mrow, kc * 32, hi, 136);
      v16h bf = fragB_glb(out_w, ncol, kc * 32, hi, 64, 64);
      acc = __builtin_amdgcn_wmma_f32_16x16x32_f16(false, a, false, bf, (short)0, acc, false, false);
    }
    float bias = out_b[ncol];
    float sk = skip1[ncol];
#pragma unroll
    for (int r = 0; r < 8; ++r) {
      int m = mtile * 16 + (hi ? 8 : 0) + r;
      int hh = wy * 8 + (m >> 3), ww = wx * 8 + (m & 7);
      size_t idx = ((size_t)b * HW + hh * Ww + ww) * 64 + ncol;
      x1[idx] = sc[idx] * sk + acc[r] + bias;
    }
  }
}

// ---------------- K3: LN2 ----------------
__global__ void k3_ln2(const float* __restrict__ x1, const float* __restrict__ g,
                       const float* __restrict__ bta, float* __restrict__ x2) {
  int lane = threadIdx.x & 31, wid = threadIdx.x >> 5;
  size_t token = (size_t)blockIdx.x * 8 + wid;
  const float* src = x1 + token * 64;
  float v0 = src[lane], v1 = src[lane + 32];
  float s  = warp_sum(v0 + v1);
  float sq = warp_sum(v0*v0 + v1*v1);
  float mu = s * (1.f/64.f), var = sq * (1.f/64.f) - mu*mu;
  float rs = rsqrtf(var + 1e-5f);
  float* dst = x2 + token * 64;
  dst[lane]      = (v0 - mu) * rs * g[lane]      + bta[lane];
  dst[lane + 32] = (v1 - mu) * rs * g[lane + 32] + bta[lane + 32];
}

// ---------------- K4: 3x3 conv 64->16 + GELU ----------------
__global__ void k4_conv1_gelu(const float* __restrict__ x2, const float* __restrict__ w1,
                              const float* __restrict__ b1, float* __restrict__ c1) {
  __shared__ float tile[3][18][64];
  int b = blockIdx.z, h = blockIdx.y, wbase = blockIdx.x * 16;
  int tid = threadIdx.x;
  for (int idx = tid; idx < 3 * 18 * 64; idx += 256) {
    int r = idx / (18 * 64);
    int remn = idx % (18 * 64);
    int wc = remn >> 6, c = remn & 63;
    int hh = h - 1 + r, ww = wbase - 1 + wc;
    float v = 0.f;
    if (hh >= 0 && hh < Hh && ww >= 0 && ww < Ww)
      v = x2[((size_t)b * HW + hh * Ww + ww) * 64 + c];
    tile[r][wc][c] = v;
  }
  __syncthreads();
  int co = tid & 15, wi = tid >> 4;
  float acc = b1[co];
  for (int kh = 0; kh < 3; ++kh)
    for (int kw = 0; kw < 3; ++kw) {
      const float* wp = w1 + (size_t)co * 576 + kh * 3 + kw;
      const float* tp = tile[kh][wi + kw];
#pragma unroll 8
      for (int c = 0; c < 64; ++c)
        acc += tp[c] * wp[c * 9];
    }
  float gl = 0.5f * acc * (1.f + erff(acc * 0.70710678f));
  c1[((size_t)b * HW + h * Ww + wbase + wi) * 16 + co] = gl;
}

// ---------------- K5: 3x3 conv 16->64 ----------------
__global__ void k5_conv2(const float* __restrict__ c1, const float* __restrict__ w2,
                         const float* __restrict__ b2, float* __restrict__ c2) {
  __shared__ float tile[3][6][16];
  int b = blockIdx.z, h = blockIdx.y, wbase = blockIdx.x * 4;
  int tid = threadIdx.x;
  for (int idx = tid; idx < 3 * 6 * 16; idx += 256) {
    int r = idx / 96, remn = idx % 96;
    int wc = remn >> 4, c = remn & 15;
    int hh = h - 1 + r, ww = wbase - 1 + wc;
    float v = 0.f;
    if (hh >= 0 && hh < Hh && ww >= 0 && ww < Ww)
      v = c1[((size_t)b * HW + hh * Ww + ww) * 16 + c];
    tile[r][wc][c] = v;
  }
  __syncthreads();
  int co = tid & 63, wi = tid >> 6;
  float acc = b2[co];
  for (int kh = 0; kh < 3; ++kh)
    for (int kw = 0; kw < 3; ++kw) {
      const float* wp = w2 + (size_t)co * 144 + kh * 3 + kw;
      const float* tp = tile[kh][wi + kw];
#pragma unroll
      for (int c = 0; c < 16; ++c)
        acc += tp[c] * wp[c * 9];
    }
  c2[((size_t)b * HW + h * Ww + wbase + wi) * 64 + co] = acc;
}

// ---------------- K6: channel attention ----------------
__global__ void k6_att(const float* __restrict__ c2, const float* __restrict__ wd,
                       const float* __restrict__ bd, const float* __restrict__ wu,
                       const float* __restrict__ bu, float* __restrict__ att) {
  __shared__ float part[4][64];
  __shared__ float meanv[64];
  __shared__ float hid[4];
  int b = blockIdx.x, tid = threadIdx.x;
  int co = tid & 63, seg = tid >> 6;
  float s = 0.f;
  const float* base = c2 + (size_t)b * HW * 64 + co;
  for (int p = seg * (HW / 4); p < (seg + 1) * (HW / 4); ++p) s += base[(size_t)p * 64];
  part[seg][co] = s;
  __syncthreads();
  if (tid < 64) meanv[tid] = (part[0][tid] + part[1][tid] + part[2][tid] + part[3][tid]) * (1.f / (float)HW);
  __syncthreads();
  if (tid < 4) {
    float a = bd[tid];
    for (int c = 0; c < 64; ++c) a += meanv[c] * wd[tid * 64 + c];
    hid[tid] = a > 0.f ? a : 0.f;
  }
  __syncthreads();
  if (tid < 64) {
    float a = bu[tid];
    for (int j = 0; j < 4; ++j) a += hid[j] * wu[tid * 4 + j];
    att[b * 64 + tid] = sigm_(a);
  }
}

// ---------------- K7: x1*skip2 + c*att, NHWC -> NCHW ----------------
__global__ void k7_out(const float* __restrict__ x1, const float* __restrict__ c2,
                       const float* __restrict__ att, const float* __restrict__ skip2,
                       float* __restrict__ out) {
  __shared__ float tile[64][65];
  int b = blockIdx.z, h = blockIdx.y, wbase = blockIdx.x * 64;
  int tid = threadIdx.x;
  int c = tid & 63, wq = tid >> 6;
  float sk = skip2[c], at = att[b * 64 + c];
  for (int wi = wq; wi < 64; wi += 4) {
    size_t idx = ((size_t)b * HW + h * Ww + wbase + wi) * 64 + c;
    tile[wi][c] = x1[idx] * sk + c2[idx] * at;
  }
  __syncthreads();
  int wi2 = tid & 63, cq = tid >> 6;
  for (int cc = cq; cc < 64; cc += 4)
    out[((size_t)(b * 64 + cc) * Hh + h) * Ww + wbase + wi2] = tile[wi2][cc];
}

extern "C" void kernel_launch(void* const* d_in, const int* in_sizes, int n_in,
                              void* d_out, int out_size, void* d_ws, size_t ws_size,
                              hipStream_t stream) {
  (void)in_sizes; (void)n_in; (void)out_size; (void)ws_size;
  const float* x       = (const float*)d_in[0];
  const float* ln1_g   = (const float*)d_in[1];
  const float* ln1_b   = (const float*)d_in[2];
  const float* in_w    = (const float*)d_in[3];
  const float* in_b    = (const float*)d_in[4];
  const float* conv_w  = (const float*)d_in[5];
  const float* conv_b  = (const float*)d_in[6];
  const float* xproj_w = (const float*)d_in[7];
  const float* dt_w    = (const float*)d_in[8];
  const float* dt_b    = (const float*)d_in[9];
  const float* A_log   = (const float*)d_in[10];
  const float* Dv      = (const float*)d_in[11];
  const float* out_w   = (const float*)d_in[12];
  const float* out_b   = (const float*)d_in[13];
  const float* skip1   = (const float*)d_in[14];
  const float* ln2_g   = (const float*)d_in[15];
  const float* ln2_b   = (const float*)d_in[16];
  const float* skip2   = (const float*)d_in[17];
  const float* cab_w1  = (const float*)d_in[18];
  const float* cab_b1  = (const float*)d_in[19];
  const float* cab_w2  = (const float*)d_in[20];
  const float* cab_b2  = (const float*)d_in[21];
  const float* ca_wd   = (const float*)d_in[22];
  const float* ca_bd   = (const float*)d_in[23];
  const float* ca_wu   = (const float*)d_in[24];
  const float* ca_bu   = (const float*)d_in[25];
  float* out = (float*)d_out;

  float* ws   = (float*)d_ws;
  float* sc   = ws;                    // [2,192,192,64]      4,718,592
  float* xin  = sc   + 4718592;        // [1152,64,128]       9,437,184
  float* zbuf = xin  + 9437184;        // [1152,64,128]       9,437,184
  float* x1   = zbuf + 9437184;        // [2,192,192,64]      4,718,592
  float* attb = x1   + 4718592;        // [2,64]
  float* x2   = xin;                   // reuse (xin dead after k2)
  float* c1   = zbuf;                  // reuse (z dead after k2) [2,192,192,16]
  float* c2   = zbuf + 1179648;        //                         [2,192,192,64]

  k0_nchw_to_nhwc<<<dim3(576, 2), 256, 0, stream>>>(x, sc);
  k1_ln_inproj<<<1152, 256, 0, stream>>>(sc, ln1_g, ln1_b, in_w, in_b, xin, zbuf);
  k2_conv_scan_outproj<<<1152, 256, 0, stream>>>(xin, zbuf, conv_w, conv_b, xproj_w,
                                                 dt_w, dt_b, A_log, Dv, out_w, out_b,
                                                 sc, skip1, x1);
  k3_ln2<<<9216, 256, 0, stream>>>(x1, ln2_g, ln2_b, x2);
  k4_conv1_gelu<<<dim3(12, 192, 2), 256, 0, stream>>>(x2, cab_w1, cab_b1, c1);
  k5_conv2<<<dim3(48, 192, 2), 256, 0, stream>>>(c1, cab_w2, cab_b2, c2);
  k6_att<<<2, 256, 0, stream>>>(c2, ca_wd, ca_bd, ca_wu, ca_bu, attb);
  k7_out<<<dim3(3, 192, 2), 256, 0, stream>>>(x1, c2, attb, skip2, out);
}
